// LlamaDifferentialAttentionBase_19988777796148
// MI455X (gfx1250) — compile-verified, matrix operand order checked by reading, not run on hardware
//
#include <hip/hip_runtime.h>
#include <math.h>
#include <stdint.h>

#define S_LEN   2048
#define HID_DIM 2048
#define NHEADS  32
#define NKVH    8
#define DHEAD   64
#define NQ      4096   // 2*HID
#define NKC     1024   // 2*D*HKV
#define NVC     512    // D*HKV
#define NQKV    5632
#define LAMBDA_INIT 0.78360576653162f

typedef __attribute__((ext_vector_type(16))) __bf16 bf16x16;
typedef __attribute__((ext_vector_type(8)))  float  f32x8;
typedef __attribute__((ext_vector_type(4)))  unsigned int uint32x4;
typedef __attribute__((ext_vector_type(8)))  int int32x8;
typedef __attribute__((ext_vector_type(4)))  int int32x4;
typedef unsigned short ushort_t;

union FragU { bf16x16 v; uint4 q[2]; };

__device__ __forceinline__ ushort_t f32_to_bf16(float f) {
  unsigned int u = __float_as_uint(f);
  u += 0x7fffu + ((u >> 16) & 1u);          // round-to-nearest-even
  return (ushort_t)(u >> 16);
}

__device__ __forceinline__ f32x8 zero8() {
  f32x8 v;
  #pragma unroll
  for (int i = 0; i < 8; i++) v[i] = 0.0f;
  return v;
}

// CDNA5 async global->LDS copy (16B), tracked by ASYNCcnt.
__device__ __forceinline__ void async_copy_b128(const ushort_t* lds_dst,
                                                const ushort_t* gsrc) {
  unsigned l = (unsigned)(uintptr_t)lds_dst;
  unsigned long long g = (unsigned long long)(uintptr_t)gsrc;
  asm volatile("global_load_async_to_lds_b128 %0, %1, off"
               :: "v"(l), "v"(g) : "memory");
}
__device__ __forceinline__ void wait_async0() {
  asm volatile("s_wait_asynccnt 0" ::: "memory");
}
__device__ __forceinline__ void wait_async6() {
  asm volatile("s_wait_asynccnt 6" ::: "memory");
}

// Tensor Data Mover: DMA a 2D bf16 tile (tile_x contiguous elems per row,
// tile_y rows, row pitch row_stride elems) from global into LDS, row-major.
// D# per ISA 8.3/8.4: group0 = {flags, lds_addr, global_addr, type=2},
// group1 = {data_size=2B, tensor dims, tile dims, dim0 stride}.
__device__ __forceinline__ void tdm_load_2d(const ushort_t* lds_dst,
                                            const ushort_t* gsrc,
                                            unsigned tile_x, unsigned tile_y,
                                            unsigned row_stride) {
  unsigned lds_addr = (unsigned)(uintptr_t)lds_dst;
  unsigned long long ga = (unsigned long long)(uintptr_t)gsrc;
  uint32x4 g0;
  g0[0] = 1u;                                          // count=1 valid descriptor
  g0[1] = lds_addr;                                    // bits 63:32 lds_addr
  g0[2] = (unsigned)(ga & 0xffffffffu);                // global_addr[31:0]
  g0[3] = (unsigned)((ga >> 32) & 0x01ffffffu) | (2u << 30);  // addr[56:32]|type=2
  int32x8 g1;
  g1[0] = (int)(1u << 16);                             // data_size=1 -> 2 bytes
  g1[1] = (int)((tile_x & 0xffffu) << 16);             // tensor_dim0 = tile_x
  g1[2] = (int)((tile_y & 0xffffu) << 16);             // tensor_dim1 = tile_y
  g1[3] = (int)(tile_x << 16);                         // tile_dim0
  g1[4] = (int)(tile_y & 0xffffu);                     // tile_dim1 (tile_dim2=0)
  g1[5] = (int)row_stride;                             // tensor_dim0_stride lo32
  g1[6] = 0;
  g1[7] = 0;
  int32x4 z4 = {0, 0, 0, 0};
#if defined(__clang_major__) && (__clang_major__ >= 23)
  int32x8 z8 = {0, 0, 0, 0, 0, 0, 0, 0};
  __builtin_amdgcn_tensor_load_to_lds(g0, g1, z4, z4, z8, 0);
#else
  __builtin_amdgcn_tensor_load_to_lds(g0, g1, z4, z4, 0);
#endif
}

// Load one 16x32 bf16 WMMA operand fragment from a row-major LDS tile.
__device__ __forceinline__ bf16x16 ldsfrag(const ushort_t* tile, int row0, int k0,
                                           int ldk, int lane) {
  int half = lane >> 4, r = lane & 15;
  const ushort_t* p = tile + (row0 + r) * ldk + k0 + half * 8;
  FragU f;
  f.q[0] = *(const uint4*)p;
  f.q[1] = *(const uint4*)(p + 16);
  return f.v;
}

#define WMMA_BF16(A, B, C) \
  __builtin_amdgcn_wmma_f32_16x16x32_bf16(false, (A), false, (B), (short)0, (C), false, false)

// ---------------------------------------------------------------- convert
__global__ void convert_f32_bf16(const float* __restrict__ in,
                                 ushort_t* __restrict__ out, int n) {
  int i = blockIdx.x * 256 + threadIdx.x;
  if (i < n) out[i] = f32_to_bf16(in[i]);
}

// ---------------------------------------------------------------- GEMM
// C[M,N] = A[M,K] * Bt[N,K]^T  (A,Bt bf16 row-major; C fp32)
// 128x256 block tile, BK=32; 8 waves in a 2x4 grid, each owning a 64x64
// sub-tile -> 16 v_wmma from 8 fragment loads per K-step.
// Double-buffered LDS via global_load_async_to_lds_b128; last step peeled.
__global__ __launch_bounds__(256) void gemm_bf16(
    const ushort_t* __restrict__ A, const ushort_t* __restrict__ Bt,
    float* __restrict__ C, int M, int N, int K) {
  __shared__ __align__(16) ushort_t As[2][128 * 32];
  __shared__ __align__(16) ushort_t Bs[2][256 * 32];
  int tid = threadIdx.x, lane = tid & 31;
  int wid = tid >> 5, wm = wid >> 2, wn = wid & 3;
  int m0 = blockIdx.y * 128, n0 = blockIdx.x * 256;

  f32x8 acc[4][4];
  #pragma unroll
  for (int i = 0; i < 4; i++)
    #pragma unroll
    for (int t = 0; t < 4; t++) acc[i][t] = zero8();

  auto fill = [&](int buf, int k0) {
    #pragma unroll
    for (int i = 0; i < 2; i++) {        // A: 128 rows * 4 chunks = 512
      int c = tid + i * 256;
      int row = c >> 2, cg = (c & 3) * 8;
      async_copy_b128(&As[buf][row * 32 + cg],
                      A + (size_t)(m0 + row) * K + k0 + cg);
    }
    #pragma unroll
    for (int i = 0; i < 4; i++) {        // B: 256 rows * 4 chunks = 1024
      int c = tid + i * 256;
      int row = c >> 2, cg = (c & 3) * 8;
      async_copy_b128(&Bs[buf][row * 32 + cg],
                      Bt + (size_t)(n0 + row) * K + k0 + cg);
    }
  };
  auto compute = [&](int buf) {
    bf16x16 af[4];
    #pragma unroll
    for (int i = 0; i < 4; i++)
      af[i] = ldsfrag(As[buf], wm * 64 + i * 16, 0, 32, lane);
    #pragma unroll
    for (int t = 0; t < 4; t++) {
      bf16x16 b = ldsfrag(Bs[buf], wn * 64 + t * 16, 0, 32, lane);
      #pragma unroll
      for (int i = 0; i < 4; i++)
        acc[i][t] = WMMA_BF16(af[i], b, acc[i][t]);
    }
  };

  int nsteps = K / 32;
  fill(0, 0);
  for (int j = 0; j < nsteps - 1; j++) {
    int buf = j & 1;
    fill(buf ^ 1, (j + 1) * 32);   // prefetch overlaps this step's WMMAs
    wait_async6();                 // oldest 6 (this buf) done; 6 stay in flight
    __syncthreads();
    compute(buf);
    __syncthreads();               // reads done before buf is refilled
  }
  wait_async0();
  __syncthreads();
  compute((nsteps - 1) & 1);

  int halfo = lane >> 4, col = lane & 15;
  #pragma unroll
  for (int i = 0; i < 4; i++)
    #pragma unroll
    for (int t = 0; t < 4; t++)
      #pragma unroll
      for (int r = 0; r < 8; r++) {
        int row = m0 + wm * 64 + i * 16 + r + halfo * 8;
        C[(size_t)row * N + n0 + wn * 64 + t * 16 + col] = acc[i][t][r];
      }
}

// ---------------------------------------------------------------- RoPE
__global__ void rope_kernel(const float* __restrict__ qkv,
                            ushort_t* __restrict__ qb, ushort_t* __restrict__ kb,
                            ushort_t* __restrict__ vb) {
  int s = blockIdx.x, tid = threadIdx.x;
  float pos = (float)s;
  const float* row = qkv + (size_t)s * NQKV;
  const float LOGTH = 0.28782313662425583f;  // ln(10000)/32

  for (int p = tid; p < 2 * NHEADS * 32; p += 256) {       // Q pairs
    int st = p >> 10, rem = p & 1023;
    int h = rem >> 5, i = rem & 31;
    int base = st * HID_DIM + h * DHEAD;
    float x0 = row[base + i], x1 = row[base + i + 32];
    float ang = pos * __expf(-(float)i * LOGTH);
    float c = cosf(ang), sn = sinf(ang);
    qb[(size_t)s * NQ + base + i]      = f32_to_bf16(x0 * c - x1 * sn);
    qb[(size_t)s * NQ + base + i + 32] = f32_to_bf16(x1 * c + x0 * sn);
  }
  for (int p = tid; p < 2 * NKVH * 32; p += 256) {         // K pairs
    int st = p >> 8, rem = p & 255;
    int h = rem >> 5, i = rem & 31;
    int base = st * 512 + h * DHEAD;
    float x0 = row[NQ + base + i], x1 = row[NQ + base + i + 32];
    float ang = pos * __expf(-(float)i * LOGTH);
    float c = cosf(ang), sn = sinf(ang);
    kb[(size_t)s * NKC + base + i]      = f32_to_bf16(x0 * c - x1 * sn);
    kb[(size_t)s * NKC + base + i + 32] = f32_to_bf16(x1 * c + x0 * sn);
  }
  for (int e = tid; e < NVC; e += 256)                     // V convert
    vb[(size_t)s * NVC + e] = f32_to_bf16(row[NQ + NKC + e]);
}

// ---------------------------------------------------------------- attention
__global__ __launch_bounds__(256) void diff_attn_kernel(
    const ushort_t* __restrict__ qb, const ushort_t* __restrict__ kb,
    const ushort_t* __restrict__ vb,
    const float* __restrict__ lq1, const float* __restrict__ lk1,
    const float* __restrict__ lq2, const float* __restrict__ lk2,
    const float* __restrict__ subw, ushort_t* __restrict__ ob) {
  __shared__ __align__(16) ushort_t Q1s[64 * 64];
  __shared__ __align__(16) ushort_t Q2s[64 * 64];
  __shared__ __align__(16) ushort_t K1s[32 * 64];
  __shared__ __align__(16) ushort_t K2s[32 * 64];
  __shared__ __align__(16) ushort_t Vts[64 * 32];   // V transposed: [d][k]
  __shared__ __align__(16) ushort_t P1s[64 * 32];
  __shared__ __align__(16) ushort_t P2s[64 * 32];
  __shared__ __align__(16) float S1s[64 * 32];
  __shared__ __align__(16) float S2s[64 * 32];
  __shared__ float m1s[64], l1s[64], m2s[64], l2s[64], c1s[64], c2s[64], red[64];

  int tid = threadIdx.x, lane = tid & 31, wid = tid >> 5;
  int wm = wid >> 1, wn = wid & 1, halfo = lane >> 4;
  int qblk = blockIdx.x, h = blockIdx.y, kvh = h >> 2;
  int qbase = qblk * 64;

  float d1 = 0.f, d2 = 0.f;
  for (int i = 0; i < DHEAD; i++) { d1 += lq1[i] * lk1[i]; d2 += lq2[i] * lk2[i]; }
  float lambda_full = __expf(d1) - __expf(d2) + LAMBDA_INIT;

  if (wid == 0) {  // persistent Q1/Q2 tiles via Tensor Data Mover
    const ushort_t* qp = qb + (size_t)qbase * NQ + h * DHEAD;
    tdm_load_2d(Q1s, qp,           DHEAD, 64, NQ);
    tdm_load_2d(Q2s, qp + HID_DIM, DHEAD, 64, NQ);
  }
  if (tid < 64) { m1s[tid] = -1e30f; m2s[tid] = -1e30f; l1s[tid] = 0.f; l2s[tid] = 0.f; }

  f32x8 o1a[2], o2a[2];
  o1a[0] = zero8(); o1a[1] = zero8(); o2a[0] = zero8(); o2a[1] = zero8();

  int njt = 2 * qblk + 2;                 // causal: KV tiles of 32
  for (int j = 0; j < njt; j++) {
    int kbase = j * 32;
    __syncthreads();
    if (wid == 0) {  // K1,K2 tiles via TDM (DMA overlaps the V transpose below)
      const ushort_t* kp = kb + (size_t)kbase * NKC + kvh * DHEAD;
      tdm_load_2d(K1s, kp,       DHEAD, 32, NKC);
      tdm_load_2d(K2s, kp + 512, DHEAD, 32, NKC);
    }
    { // V transposed manually (TDM has no transpose)
      int r = tid >> 3, cg = (tid & 7) * 8;
      uint4 vq = *(const uint4*)(vb + (size_t)(kbase + r) * NVC + kvh * DHEAD + cg);
      const ushort_t* ve = (const ushort_t*)&vq;
      #pragma unroll
      for (int i = 0; i < 8; i++) Vts[(cg + i) * 32 + r] = ve[i];
    }
    if (wid == 0) __builtin_amdgcn_s_wait_tensorcnt(0);
    __syncthreads();

    // scores: wave covers rows wm*16..+15, cols wn*16..+15 of 64x32
    f32x8 s1 = zero8(), s2 = zero8();
    #pragma unroll
    for (int dk = 0; dk < 64; dk += 32) {
      bf16x16 a1 = ldsfrag(Q1s, wm * 16, dk, 64, lane);
      bf16x16 b1 = ldsfrag(K1s, wn * 16, dk, 64, lane);
      s1 = WMMA_BF16(a1, b1, s1);
      bf16x16 a2 = ldsfrag(Q2s, wm * 16, dk, 64, lane);
      bf16x16 b2 = ldsfrag(K2s, wn * 16, dk, 64, lane);
      s2 = WMMA_BF16(a2, b2, s2);
    }
    {
      int col = wn * 16 + (lane & 15);
      #pragma unroll
      for (int r = 0; r < 8; r++) {
        int row = wm * 16 + r + halfo * 8;
        float v1 = s1[r] * 0.125f, v2 = s2[r] * 0.125f;
        if (kbase + col > qbase + row) { v1 = -1e30f; v2 = -1e30f; }
        S1s[row * 32 + col] = v1;
        S2s[row * 32 + col] = v2;
      }
    }
    __syncthreads();

    if (tid < 64) {  // per-row online softmax, both streams
      float mo = m1s[tid], mx = mo;
      for (int c = 0; c < 32; c++) mx = fmaxf(mx, S1s[tid * 32 + c]);
      float corr = __expf(mo - mx);
      float l = l1s[tid] * corr;
      for (int c = 0; c < 32; c++) {
        float pv = __expf(S1s[tid * 32 + c] - mx);
        l += pv;
        P1s[tid * 32 + c] = f32_to_bf16(pv);
      }
      m1s[tid] = mx; l1s[tid] = l; c1s[tid] = corr;

      mo = m2s[tid]; mx = mo;
      for (int c = 0; c < 32; c++) mx = fmaxf(mx, S2s[tid * 32 + c]);
      corr = __expf(mo - mx);
      l = l2s[tid] * corr;
      for (int c = 0; c < 32; c++) {
        float pv = __expf(S2s[tid * 32 + c] - mx);
        l += pv;
        P2s[tid * 32 + c] = f32_to_bf16(pv);
      }
      m2s[tid] = mx; l2s[tid] = l; c2s[tid] = corr;
    }
    __syncthreads();

    // rescale accumulators + P*V (K=32 -> one WMMA per 16x16 out tile)
    bf16x16 pa1 = ldsfrag(P1s, wm * 16, 0, 32, lane);
    bf16x16 pa2 = ldsfrag(P2s, wm * 16, 0, 32, lane);
    #pragma unroll
    for (int t = 0; t < 2; t++) {
      #pragma unroll
      for (int r = 0; r < 8; r++) {
        int row = wm * 16 + r + halfo * 8;
        o1a[t][r] *= c1s[row];
        o2a[t][r] *= c2s[row];
      }
      bf16x16 vf = ldsfrag(Vts, wn * 32 + t * 16, 0, 32, lane);
      o1a[t] = WMMA_BF16(pa1, vf, o1a[t]);
      o2a[t] = WMMA_BF16(pa2, vf, o2a[t]);
    }
  }

  __syncthreads();
  if (tid < 64) red[tid] = 0.f;
  __syncthreads();
  float onew[2][8];
  #pragma unroll
  for (int t = 0; t < 2; t++)
    #pragma unroll
    for (int r = 0; r < 8; r++) {
      int row = wm * 16 + r + halfo * 8;
      float o = o1a[t][r] / l1s[row] - lambda_full * (o2a[t][r] / l2s[row]);
      onew[t][r] = o;
      atomicAdd(&red[row], o * o);   // ds_add_f32
    }
  __syncthreads();
  if (tid < 64) red[tid] = rsqrtf(red[tid] * (1.0f / 64.0f) + 1e-6f);
  __syncthreads();
  #pragma unroll
  for (int t = 0; t < 2; t++)
    #pragma unroll
    for (int r = 0; r < 8; r++) {
      int row = wm * 16 + r + halfo * 8;
      int d = wn * 32 + t * 16 + (lane & 15);
      float val = onew[t][r] * red[row] * subw[d] * (1.0f - LAMBDA_INIT);
      ob[(size_t)(qbase + row) * HID_DIM + h * DHEAD + d] = f32_to_bf16(val);
    }
}

// ---------------------------------------------------------------- launch
extern "C" void kernel_launch(void* const* d_in, const int* in_sizes, int n_in,
                              void* d_out, int out_size, void* d_ws, size_t ws_size,
                              hipStream_t stream) {
  (void)in_sizes; (void)n_in; (void)out_size; (void)ws_size;
  const float* hidden = (const float*)d_in[0];
  const float* Wq = (const float*)d_in[1];
  const float* Wk = (const float*)d_in[2];
  const float* Wv = (const float*)d_in[3];
  const float* Wo = (const float*)d_in[4];
  const float* lq1 = (const float*)d_in[5];
  const float* lk1 = (const float*)d_in[6];
  const float* lq2 = (const float*)d_in[7];
  const float* lk2 = (const float*)d_in[8];
  const float* subw = (const float*)d_in[9];
  // d_in[10] = position_ids (arange, recomputed in-kernel)

  char* ws = (char*)d_ws;
  size_t off = 0;
  ushort_t* hidb  = (ushort_t*)(ws + off); off += (size_t)S_LEN * HID_DIM * 2;
  ushort_t* wqkvb = (ushort_t*)(ws + off); off += (size_t)NQKV * HID_DIM * 2;
  ushort_t* wob   = (ushort_t*)(ws + off); off += (size_t)HID_DIM * HID_DIM * 2;
  float*    qkvf  = (float*)(ws + off);    off += (size_t)S_LEN * NQKV * 4;
  ushort_t* qbp   = (ushort_t*)(ws + off); off += (size_t)S_LEN * NQ * 2;
  ushort_t* kbp   = (ushort_t*)(ws + off); off += (size_t)S_LEN * NKC * 2;
  ushort_t* vbp   = (ushort_t*)(ws + off); off += (size_t)S_LEN * NVC * 2;
  ushort_t* obp   = (ushort_t*)(ws + off); off += (size_t)S_LEN * HID_DIM * 2;

  int n;
  n = S_LEN * HID_DIM;
  convert_f32_bf16<<<(n + 255) / 256, 256, 0, stream>>>(hidden, hidb, n);
  n = NQ * HID_DIM;
  convert_f32_bf16<<<(n + 255) / 256, 256, 0, stream>>>(Wq, wqkvb, n);
  n = NKC * HID_DIM;
  convert_f32_bf16<<<(n + 255) / 256, 256, 0, stream>>>(Wk, wqkvb + (size_t)NQ * HID_DIM, n);
  n = NVC * HID_DIM;
  convert_f32_bf16<<<(n + 255) / 256, 256, 0, stream>>>(Wv, wqkvb + (size_t)(NQ + NKC) * HID_DIM, n);
  n = HID_DIM * HID_DIM;
  convert_f32_bf16<<<(n + 255) / 256, 256, 0, stream>>>(Wo, wob, n);

  // fused QKV projection: [2048 x 5632] = hidden * [Wq|Wk|Wv]^T
  gemm_bf16<<<dim3(NQKV / 256, S_LEN / 128), 256, 0, stream>>>(hidb, wqkvb, qkvf,
                                                               S_LEN, NQKV, HID_DIM);
  rope_kernel<<<S_LEN, 256, 0, stream>>>(qkvf, qbp, kbp, vbp);

  diff_attn_kernel<<<dim3(S_LEN / 64, NHEADS), 256, 0, stream>>>(
      qbp, kbp, vbp, lq1, lk1, lq2, lk2, subw, obp);

  // output projection -> fp32 d_out
  gemm_bf16<<<dim3(HID_DIM / 256, S_LEN / 128), 256, 0, stream>>>(
      obp, wob, (float*)d_out, S_LEN, HID_DIM, HID_DIM);
}
